// MultiHeadGAT_68573447848301
// MI455X (gfx1250) — matrix operand order repaired and checked by reading, not compiled, not run on hardware
//
#include <hip/hip_runtime.h>
#include <hip/hip_bf16.h>
#include <math.h>

typedef _Float16 v16h __attribute__((ext_vector_type(16)));
typedef _Float16 v8h  __attribute__((ext_vector_type(8)));
typedef _Float16 v4h  __attribute__((ext_vector_type(4)));
typedef float    v8f  __attribute__((ext_vector_type(8)));

#define FDIM 128
#define NHEAD 4
#define SLOPE 0.2f
#define LPAD (FDIM + 8)     // f16 LDS row stride: 136 elems = 272 B (16B aligned)
#define RT 4                // row tiles (of 16) per block -> 64 rows/block

// ------------------------------ helpers ------------------------------------
__device__ __forceinline__ float leaky(float t) { return t > 0.f ? t : SLOPE * t; }

// monotone float <-> uint key for atomicMax on floats
__device__ __forceinline__ unsigned f2key(float f) {
  unsigned b = __float_as_uint(f);
  return (b & 0x80000000u) ? ~b : (b | 0x80000000u);
}
__device__ __forceinline__ float key2f(unsigned k) {
  unsigned b = (k & 0x80000000u) ? (k & 0x7FFFFFFFu) : ~k;
  return __uint_as_float(b);
}

__device__ __forceinline__ void edge_sd(const int* __restrict__ ei, long long i,
                                        long long E, int& s, int& d) {
  if (i < E) { s = ei[i]; d = ei[E + i]; }
  else       { s = (int)(i - E); d = s; }   // appended self-loops
}

__global__ void fill_f32(float* __restrict__ p, long long n, float v) {
  long long i = (long long)blockIdx.x * blockDim.x + threadIdx.x;
  if (i < n) p[i] = v;
}

// ------------------------- WMMA GEMM: H = X * W ----------------------------
// X:[N,128] f32, W:[128,128] f32, H:[N,128] f32.
// Block = 64 rows (4 row-tiles), 8 waves; wave w owns column tile w.
// W staged transposed in LDS so every fragment is two contiguous 16B runs.
__global__ __launch_bounds__(256)
void gat_gemm_wmma(const float* __restrict__ X, const float* __restrict__ W,
                   float* __restrict__ H, int N) {
  __shared__ _Float16 sA[RT * 16][LPAD];    // sA[m][k]
  __shared__ _Float16 sBt[FDIM][LPAD];      // sBt[n][k] = W[k][n]
  const int tid = threadIdx.x;
  const int r0  = blockIdx.x * (RT * 16);

  // stage A tile (64 x 128) as f16, float4 loads -> packed v4h stores
  #pragma unroll
  for (int j = 0; j < 8; ++j) {
    int vi  = tid + 256 * j;                // 0..2047 (float4 units)
    int li  = vi * 4;
    int row = li >> 7, col = li & 127;
    int gr  = r0 + row;
    float4 v = make_float4(0.f, 0.f, 0.f, 0.f);
    if (gr < N) v = *(const float4*)&X[(long long)gr * FDIM + col];
    v4h h4 = { (_Float16)v.x, (_Float16)v.y, (_Float16)v.z, (_Float16)v.w };
    *(v4h*)&sA[row][col] = h4;
  }
  // stage W transposed (16384 elems): float4 row loads, 16b scatter stores
  #pragma unroll
  for (int j = 0; j < 16; ++j) {
    int vi  = tid + 256 * j;                // 0..4095 (float4 units)
    int li  = vi * 4;
    int row = li >> 7, col = li & 127;
    float4 v = *(const float4*)&W[(long long)row * FDIM + col];
    sBt[col + 0][row] = (_Float16)v.x;
    sBt[col + 1][row] = (_Float16)v.y;
    sBt[col + 2][row] = (_Float16)v.z;
    sBt[col + 3][row] = (_Float16)v.w;
  }
  __syncthreads();

  const int wave = tid >> 5;                // column tile 0..7
  const int lane = tid & 31;
  const int ml   = lane & 15;
  const int lh   = lane >> 4;
  const int colb = wave * 16 + ml;

  v8f acc[RT] = {};
  #pragma unroll
  for (int kk = 0; kk < FDIM; kk += 32) {
    // B 32x16 f16 fragment: lanes0-15 slot i -> K=kk+i; lanes16-31 -> K=kk+16+i
    v8h b0 = *(const v8h*)&sBt[colb][kk + lh * 16];
    v8h b1 = *(const v8h*)&sBt[colb][kk + lh * 16 + 8];
    v16h b = __builtin_shufflevector(b0, b1, 0, 1, 2, 3, 4, 5, 6, 7,
                                     8, 9, 10, 11, 12, 13, 14, 15);
    #pragma unroll
    for (int rt = 0; rt < RT; ++rt) {
      // A 16x32 f16 fragment: lanes0-15 slot i -> K=kk+i; lanes16-31 -> K=kk+16+i
      const int mrow = rt * 16 + ml;
      v8h a0 = *(const v8h*)&sA[mrow][kk + lh * 16];
      v8h a1 = *(const v8h*)&sA[mrow][kk + lh * 16 + 8];
      v16h a = __builtin_shufflevector(a0, a1, 0, 1, 2, 3, 4, 5, 6, 7,
                                       8, 9, 10, 11, 12, 13, 14, 15);
      acc[rt] = __builtin_amdgcn_wmma_f32_16x16x32_f16(false, a, false, b,
                                                       (short)0, acc[rt],
                                                       false, false);
    }
  }
  // C/D layout: VGPR j -> M = j (lanes0-15) / j+8 (lanes16-31), N = lane&15
  #pragma unroll
  for (int rt = 0; rt < RT; ++rt) {
    #pragma unroll
    for (int j = 0; j < 8; ++j) {
      int gr = r0 + rt * 16 + j + lh * 8;
      if (gr < N) H[(long long)gr * FDIM + colb] = acc[rt][j];
    }
  }
}

// ------------------- per-node attention logits (wave/node) -----------------
__global__ __launch_bounds__(256)
void gat_alpha(const float* __restrict__ H, const float* __restrict__ a_src,
               const float* __restrict__ a_dst, float* __restrict__ asrc,
               float* __restrict__ adst, int N) {
  int node = blockIdx.x * 8 + (threadIdx.x >> 5);
  if (node >= N) return;
  int lane = threadIdx.x & 31;
  float s[NHEAD], d[NHEAD];
  #pragma unroll
  for (int k = 0; k < NHEAD; ++k) {
    float v = H[(long long)node * FDIM + k * 32 + lane];
    s[k] = v * a_src[k * 32 + lane];
    d[k] = v * a_dst[k * 32 + lane];
  }
  #pragma unroll
  for (int k = 0; k < NHEAD; ++k) {
    #pragma unroll
    for (int off = 16; off; off >>= 1) {
      s[k] += __shfl_xor(s[k], off, 32);
      d[k] += __shfl_xor(d[k], off, 32);
    }
  }
  if (lane == 0) {
    #pragma unroll
    for (int k = 0; k < NHEAD; ++k) {
      asrc[(long long)node * NHEAD + k] = s[k];
      adst[(long long)node * NHEAD + k] = d[k];
    }
  }
}

// ------------------------ segment max over edges ---------------------------
__global__ void gat_edge_max(const int* __restrict__ ei, long long E, long long Etot,
                             const float* __restrict__ asrc, const float* __restrict__ adst,
                             unsigned* __restrict__ emaxK) {
  long long gid = (long long)blockIdx.x * blockDim.x + threadIdx.x;
  if (gid >= Etot * NHEAD) return;
  long long edge = gid >> 2;
  int k = (int)(gid & 3);
  int s, d; edge_sd(ei, edge, E, s, d);
  float e = leaky(asrc[(long long)s * NHEAD + k] + adst[(long long)d * NHEAD + k]);
  atomicMax(&emaxK[(long long)d * NHEAD + k], f2key(e));
}

// ------------------------ segment sum of exp -------------------------------
__global__ void gat_edge_sum(const int* __restrict__ ei, long long E, long long Etot,
                             const float* __restrict__ asrc, const float* __restrict__ adst,
                             const unsigned* __restrict__ emaxK, float* __restrict__ denom) {
  long long gid = (long long)blockIdx.x * blockDim.x + threadIdx.x;
  if (gid >= Etot * NHEAD) return;
  long long edge = gid >> 2;
  int k = (int)(gid & 3);
  int s, d; edge_sd(ei, edge, E, s, d);
  float e = leaky(asrc[(long long)s * NHEAD + k] + adst[(long long)d * NHEAD + k]);
  unsigned key = emaxK[(long long)d * NHEAD + k];
  float em = (key == 0u) ? 0.f : key2f(key);
  atomicAdd(&denom[(long long)d * NHEAD + k], __expf(e - em));
}

// -------------------- weighted scatter-aggregate (wave/edge) ---------------
__global__ __launch_bounds__(256)
void gat_edge_agg(const int* __restrict__ ei, long long E, long long Etot,
                  const float* __restrict__ H, const float* __restrict__ asrc,
                  const float* __restrict__ adst, const unsigned* __restrict__ emaxK,
                  const float* __restrict__ denom, float* __restrict__ agg) {
  long long edge = (long long)blockIdx.x * 8 + (threadIdx.x >> 5);
  if (edge >= Etot) return;
  int lane = threadIdx.x & 31;
  int s, d; edge_sd(ei, edge, E, s, d);
  float al[NHEAD];
  #pragma unroll
  for (int k = 0; k < NHEAD; ++k) {
    float e = leaky(asrc[(long long)s * NHEAD + k] + adst[(long long)d * NHEAD + k]);
    unsigned key = emaxK[(long long)d * NHEAD + k];
    float em = (key == 0u) ? 0.f : key2f(key);
    float dn = denom[(long long)d * NHEAD + k];
    al[k] = __expf(e - em) / (dn + 1e-16f);
  }
  #pragma unroll
  for (int k = 0; k < NHEAD; ++k) {
    float v = H[(long long)s * FDIM + k * 32 + lane];
    atomicAdd(&agg[(long long)d * FDIM + k * 32 + lane], v * al[k]);
  }
}

// ----------------------- bias + ELU (between layers) -----------------------
__global__ void gat_bias_elu(const float* __restrict__ agg, const float* __restrict__ b,
                             float* __restrict__ dst, long long n) {
  long long i = (long long)blockIdx.x * blockDim.x + threadIdx.x;
  if (i >= n) return;
  int c = (int)(i & (FDIM - 1));
  float v = agg[i] + b[c];
  dst[i] = v > 0.f ? v : expm1f(v);
}

// -------------- final bias + per-head split writes into d_out --------------
__global__ void gat_finalize(float* __restrict__ out0, const float* __restrict__ b,
                             float* __restrict__ outh, long long N) {
  long long i = (long long)blockIdx.x * blockDim.x + threadIdx.x;
  if (i >= N * FDIM) return;
  long long n = i >> 7;
  int c = (int)(i & (FDIM - 1));
  float v = out0[i] + b[c];
  out0[i] = v;
  int hd = c >> 5, cc = c & 31;
  outh[(long long)hd * N * 32 + n * 32 + cc] = v;
}

// ------------------------------- launcher ----------------------------------
extern "C" void kernel_launch(void* const* d_in, const int* in_sizes, int n_in,
                              void* d_out, int out_size, void* d_ws, size_t ws_size,
                              hipStream_t stream) {
  const float* x   = (const float*)d_in[0];
  const int*   ei  = (const int*)d_in[1];
  const float* W1  = (const float*)d_in[2];
  const float* as1 = (const float*)d_in[3];
  const float* ad1 = (const float*)d_in[4];
  const float* b1  = (const float*)d_in[5];
  const float* W2  = (const float*)d_in[6];
  const float* as2 = (const float*)d_in[7];
  const float* ad2 = (const float*)d_in[8];
  const float* b2  = (const float*)d_in[9];

  const long long N    = in_sizes[0] / FDIM;
  const long long E    = in_sizes[1] / 2;
  const long long Etot = E + N;

  float*    F0    = (float*)d_ws;                 // h1, then ELU features
  float*    F1    = F0 + N * FDIM;                // agg1, then h2
  float*    asrc  = F1 + N * FDIM;
  float*    adst  = asrc + N * NHEAD;
  unsigned* emaxK = (unsigned*)(adst + N * NHEAD);
  float*    denom = (float*)(emaxK + N * NHEAD);
  float*    out0  = (float*)d_out;                // [N,128]
  float*    outh  = out0 + N * FDIM;              // 4 x [N,32]

  const dim3 blk(256);
  const int gGemm  = (int)((N + RT * 16 - 1) / (RT * 16));
  const int gNode  = (int)((N + 7) / 8);
  const int gEdge4 = (int)((Etot * NHEAD + 255) / 256);
  const int gEdgeW = (int)((Etot + 7) / 8);
  const long long nNF = N * FDIM;
  const long long nNH = N * NHEAD;
  const int gNF = (int)((nNF + 255) / 256);
  const int gNH = (int)((nNH + 255) / 256);

  // ---------------- layer 1 ----------------
  gat_gemm_wmma<<<gGemm, blk, 0, stream>>>(x, W1, F0, (int)N);
  gat_alpha<<<gNode, blk, 0, stream>>>(F0, as1, ad1, asrc, adst, (int)N);
  fill_f32<<<gNH, blk, 0, stream>>>((float*)emaxK, nNH, 0.f);   // key 0 == -inf
  fill_f32<<<gNH, blk, 0, stream>>>(denom, nNH, 0.f);
  fill_f32<<<gNF, blk, 0, stream>>>(F1, nNF, 0.f);
  gat_edge_max<<<gEdge4, blk, 0, stream>>>(ei, E, Etot, asrc, adst, emaxK);
  gat_edge_sum<<<gEdge4, blk, 0, stream>>>(ei, E, Etot, asrc, adst, emaxK, denom);
  gat_edge_agg<<<gEdgeW, blk, 0, stream>>>(ei, E, Etot, F0, asrc, adst, emaxK, denom, F1);
  gat_bias_elu<<<gNF, blk, 0, stream>>>(F1, b1, F0, nNF);       // F0 <- ELU(agg1+b1)

  // ---------------- layer 2 ----------------
  gat_gemm_wmma<<<gGemm, blk, 0, stream>>>(F0, W2, F1, (int)N); // F1 <- h2
  gat_alpha<<<gNode, blk, 0, stream>>>(F1, as2, ad2, asrc, adst, (int)N);
  fill_f32<<<gNH, blk, 0, stream>>>((float*)emaxK, nNH, 0.f);
  fill_f32<<<gNH, blk, 0, stream>>>(denom, nNH, 0.f);
  fill_f32<<<gNF, blk, 0, stream>>>(out0, nNF, 0.f);
  gat_edge_max<<<gEdge4, blk, 0, stream>>>(ei, E, Etot, asrc, adst, emaxK);
  gat_edge_sum<<<gEdge4, blk, 0, stream>>>(ei, E, Etot, asrc, adst, emaxK, denom);
  gat_edge_agg<<<gEdgeW, blk, 0, stream>>>(ei, E, Etot, F1, asrc, adst, emaxK, denom, out0);
  gat_finalize<<<gNF, blk, 0, stream>>>(out0, b2, outh, N);
}
